// ImplicitModel_79912161509967
// MI455X (gfx1250) — compile-verified, matrix operand order checked by reading, not run on hardware
//
#include <hip/hip_runtime.h>
#include <math.h>

// ---------------------------------------------------------------------------
// ImplicitModel (DEQ fixed-point) on MI455X / gfx1250, wave32 + WMMA bf16.
//   Bu = obs @ B_T                         (f32, loop invariant)
//   x = tanh(Bu); 29x: x = tanh(x@A_T+Bu)  (bf16 activations, f32-accum WMMA)
//   y = x @ C_T + obs @ D_T                (f32 out)
// GEMM: 128x128 block tile, 8 waves (64x32 each, 4x2 WMMA accums), BK=64,
// double-buffered LDS filled by GLOBAL_LOAD_ASYNC_TO_LDS_B128 (ASYNCcnt),
// one s_wait_asynccnt + one barrier per K-step.
// ---------------------------------------------------------------------------

typedef __attribute__((ext_vector_type(16))) __bf16 v16bf;
typedef __attribute__((ext_vector_type(8)))  float  v8f;
typedef int v4i __attribute__((vector_size(16)));   // matches builtin's V4i

#define BM 128
#define BN 128
#define BK 64
#define LDSS 72   // 64 + 8 bf16 pad: 144B row stride (16B aligned), bank-spread

union Frag { v16bf v; uint4 q[2]; };

__device__ __forceinline__ void async_copy_b128(const __bf16* g, __bf16* l) {
#if __has_builtin(__builtin_amdgcn_global_load_async_to_lds_b128)
    __builtin_amdgcn_global_load_async_to_lds_b128(
        (__attribute__((address_space(1))) v4i*)g,
        (__attribute__((address_space(3))) v4i*)l, 0, 0);
#else
    asm volatile("global_load_async_to_lds_b128 %0, %1, off"
                 :: "v"((__attribute__((address_space(3))) void*)l),
                    "v"((__attribute__((address_space(1))) const void*)g)
                 : "memory");
#endif
}

__device__ __forceinline__ void wait_async0() {
#if __has_builtin(__builtin_amdgcn_s_wait_asynccnt)
    __builtin_amdgcn_s_wait_asynccnt(0);
#else
    asm volatile("s_wait_asynccnt 0x0" ::: "memory");
#endif
}

// out = act( X[M,K](bf16,row-major) @ Wt[N,K](bf16)^T + bias[M,N](f32) )
template<bool ADD_BIAS, bool TANH, bool OUT_BF16>
__global__ __launch_bounds__(256) void gemm_bf16_wmma(
    const __bf16* __restrict__ X, const __bf16* __restrict__ Wt,
    const float* __restrict__ bias,
    __bf16* __restrict__ outb, float* __restrict__ outf,
    int M, int N, int K)
{
    __shared__ __bf16 sX[2][BM * LDSS];
    __shared__ __bf16 sW[2][BN * LDSS];

    const int tid  = threadIdx.x;
    const int lane = tid & 31;
    const int wave = tid >> 5;
    const int wm   = wave & 1;   // 2 wave-rows  x 64 rows
    const int wn   = wave >> 1;  // 4 wave-cols  x 32 cols
    const int bm   = blockIdx.y * BM;
    const int bn   = blockIdx.x * BN;

    // async global->LDS: 128x64 bf16 tile = 1024 x 16B chunks, 4/thread/operand
    // chunk c: row = c>>3, col = (c&7)*8
    auto issue_tile = [&](int buf, int k0) {
        #pragma unroll
        for (int i = 0; i < 4; ++i) {
            const int c = tid + i * 256;
            const int r = c >> 3, col = (c & 7) * 8;
            async_copy_b128(X  + (size_t)(bm + r) * K + k0 + col,
                            &sX[buf][r * LDSS + col]);
            async_copy_b128(Wt + (size_t)(bn + r) * K + k0 + col,
                            &sW[buf][r * LDSS + col]);
        }
    };

    v8f acc[4][2];
    #pragma unroll
    for (int m = 0; m < 4; ++m)
        #pragma unroll
        for (int n = 0; n < 2; ++n)
            acc[m][n] = (v8f){0.f,0.f,0.f,0.f,0.f,0.f,0.f,0.f};

    issue_tile(0, 0);

    const int rbase = lane & 15;
    const int kbase = (lane >> 4) * 8;

    int buf = 0;
    for (int k0 = 0; k0 < K; k0 += BK) {
        wait_async0();       // my async fills of sX/sW[buf] are in LDS
        __syncthreads();     // everyone's fills landed; reads of buf^1 done

        if (k0 + BK < K) issue_tile(buf ^ 1, k0 + BK);

        #pragma unroll
        for (int kk = 0; kk < BK; kk += 32) {
            Frag a[4], b[2];
            #pragma unroll
            for (int m = 0; m < 4; ++m) {
                const __bf16* p =
                    &sX[buf][(wm * 64 + m * 16 + rbase) * LDSS + kk + kbase];
                a[m].q[0] = *(const uint4*)(p);
                a[m].q[1] = *(const uint4*)(p + 16);
            }
            #pragma unroll
            for (int n = 0; n < 2; ++n) {
                const __bf16* p =
                    &sW[buf][(wn * 32 + n * 16 + rbase) * LDSS + kk + kbase];
                b[n].q[0] = *(const uint4*)(p);
                b[n].q[1] = *(const uint4*)(p + 16);
            }
            #pragma unroll
            for (int m = 0; m < 4; ++m)
                #pragma unroll
                for (int n = 0; n < 2; ++n)
                    acc[m][n] = __builtin_amdgcn_wmma_f32_16x16x32_bf16(
                        false, a[m].v, false, b[n].v,
                        (short)0, acc[m][n], false, false);
        }
        buf ^= 1;
    }

    // epilogue: C/D layout (ISA 7.12.2): lane -> col = lane%16,
    //           VGPR r -> row = r + 8*(lane/16)
    const int colw  = bn + wn * 32 + (lane & 15);
    const int rowOf = (lane >> 4) * 8;
    #pragma unroll
    for (int m = 0; m < 4; ++m) {
        #pragma unroll
        for (int n = 0; n < 2; ++n) {
            const int cg = colw + n * 16;
            #pragma unroll
            for (int r = 0; r < 8; ++r) {
                const int rg = bm + wm * 64 + m * 16 + rowOf + r;
                float v = acc[m][n][r];
                if (ADD_BIAS) v += bias[(size_t)rg * N + cg];
                if (TANH)     v = tanhf(v);
                if (OUT_BF16) outb[(size_t)rg * N + cg] = (__bf16)v;
                else          outf[(size_t)rg * N + cg] = v;
            }
        }
    }
}

// elementwise f32 -> bf16
__global__ void cvt_bf16_kernel(const float* __restrict__ in,
                                __bf16* __restrict__ out, int n) {
    int i = blockIdx.x * blockDim.x + threadIdx.x;
    if (i < n) out[i] = (__bf16)in[i];
}

// in: [K][N] f32 row-major  ->  out: [N][K] bf16 (N-major)
__global__ void transpose_cvt_kernel(const float* __restrict__ in,
                                     __bf16* __restrict__ out, int K, int N) {
    int i = blockIdx.x * blockDim.x + threadIdx.x;
    if (i < N * K) {
        int n = i / K, k = i - n * K;
        out[i] = (__bf16)in[(size_t)k * N + n];
    }
}

// x = tanh(Bu)  (iteration 1, x0 == 0)
__global__ void tanh_cvt_kernel(const float* __restrict__ in,
                                __bf16* __restrict__ out, int n) {
    int i = blockIdx.x * blockDim.x + threadIdx.x;
    if (i < n) out[i] = (__bf16)tanhf(in[i]);
}

extern "C" void kernel_launch(void* const* d_in, const int* in_sizes, int n_in,
                              void* d_out, int out_size, void* d_ws, size_t ws_size,
                              hipStream_t stream) {
    (void)in_sizes; (void)n_in; (void)out_size; (void)ws_size;
    const int BATCH = 8192, IN = 512, ST = 1024, OUT = 512, ITERS = 30;

    const float* obs = (const float*)d_in[0]; // [BATCH, IN]
    const float* A_T = (const float*)d_in[1]; // [ST, ST]
    const float* B_T = (const float*)d_in[2]; // [IN, ST]
    const float* C_T = (const float*)d_in[3]; // [ST, OUT]
    const float* D_T = (const float*)d_in[4]; // [IN, OUT]
    float* y = (float*)d_out;                 // [BATCH, OUT]

    char* w = (char*)d_ws;
    size_t off = 0;
    auto alloc = [&](size_t bytes) { void* p = w + off; off += (bytes + 255) & ~(size_t)255; return p; };
    __bf16* obs_bf = (__bf16*)alloc((size_t)BATCH * IN * 2);
    __bf16* Wb     = (__bf16*)alloc((size_t)ST * IN * 2);     // [ST][IN]
    __bf16* Wa     = (__bf16*)alloc((size_t)ST * ST * 2);     // [ST][ST]
    __bf16* Wc     = (__bf16*)alloc((size_t)OUT * ST * 2);    // [OUT][ST]
    __bf16* Wd     = (__bf16*)alloc((size_t)OUT * IN * 2);    // [OUT][IN]
    float*  Bu     = (float*) alloc((size_t)BATCH * ST * 4);
    __bf16* X0     = (__bf16*)alloc((size_t)BATCH * ST * 2);
    __bf16* X1     = (__bf16*)alloc((size_t)BATCH * ST * 2);
    float*  tmp    = (float*) alloc((size_t)BATCH * OUT * 4); // obs @ D_T

    const int TPB = 256;
    // prologue: converts + weight transposes (N-major bf16)
    cvt_bf16_kernel<<<(BATCH * IN + TPB - 1) / TPB, TPB, 0, stream>>>(obs, obs_bf, BATCH * IN);
    transpose_cvt_kernel<<<(ST * IN  + TPB - 1) / TPB, TPB, 0, stream>>>(B_T, Wb, IN, ST);
    transpose_cvt_kernel<<<(ST * ST  + TPB - 1) / TPB, TPB, 0, stream>>>(A_T, Wa, ST, ST);
    transpose_cvt_kernel<<<(OUT * ST + TPB - 1) / TPB, TPB, 0, stream>>>(C_T, Wc, ST, OUT);
    transpose_cvt_kernel<<<(OUT * IN + TPB - 1) / TPB, TPB, 0, stream>>>(D_T, Wd, IN, OUT);

    // Bu = obs @ B_T   (f32 out)
    gemm_bf16_wmma<false, false, false><<<dim3(ST / BN, BATCH / BM), TPB, 0, stream>>>(
        obs_bf, Wb, nullptr, nullptr, Bu, BATCH, ST, IN);

    // iter 1: x = tanh(Bu)
    tanh_cvt_kernel<<<(BATCH * ST + TPB - 1) / TPB, TPB, 0, stream>>>(Bu, X0, BATCH * ST);

    // iters 2..30: x = tanh(x @ A_T + Bu)
    __bf16* cur = X0;
    __bf16* nxt = X1;
    for (int it = 1; it < ITERS; ++it) {
        gemm_bf16_wmma<true, true, true><<<dim3(ST / BN, BATCH / BM), TPB, 0, stream>>>(
            cur, Wa, Bu, nxt, nullptr, BATCH, ST, ST);
        __bf16* t = cur; cur = nxt; nxt = t;
    }

    // tmp = obs @ D_T  (f32)
    gemm_bf16_wmma<false, false, false><<<dim3(OUT / BN, BATCH / BM), TPB, 0, stream>>>(
        obs_bf, Wd, nullptr, nullptr, tmp, BATCH, OUT, IN);

    // y = x @ C_T + tmp  (f32 out)
    gemm_bf16_wmma<true, false, false><<<dim3(OUT / BN, BATCH / BM), TPB, 0, stream>>>(
        cur, Wc, tmp, nullptr, y, BATCH, OUT, ST);
}
